// unit_gcn_69071664054813
// MI455X (gfx1250) — matrix operand order repaired and checked by reading, not compiled, not run on hardware
//
#include <hip/hip_runtime.h>
#include <hip/hip_bf16.h>
#include <math.h>

// ---------------------------------------------------------------------------
// unit_gcn for MI455X (gfx1250): all GEMMs via v_wmma_f32_16x16x32_f16
// ---------------------------------------------------------------------------

typedef __attribute__((ext_vector_type(16))) _Float16 v16h;
typedef __attribute__((ext_vector_type(8)))  _Float16 v8h;
typedef __attribute__((ext_vector_type(8)))  float    v8f;

#define N_  64
#define C_  64
#define T_  256
#define V_  25
#define S_  3
#define TV_  (T_ * V_)        // 6400
#define CTV_ (C_ * TV_)       // 409600

// ---- workspace layout (bytes) ----
static constexpr size_t WS_AIT = 0;                                  // f16 AiT [S][N][32][32]
static constexpr size_t SZ_AIT = (size_t)S_ * N_ * 32 * 32 * 2;      // 393216
static constexpr size_t WS_ST  = WS_AIT + SZ_AIT;                    // f32 S_T [N*C][32]
static constexpr size_t SZ_ST  = (size_t)N_ * C_ * 32 * 4;
static constexpr size_t WS_GS  = WS_ST + SZ_ST;                      // f32 gs [N][32]
static constexpr size_t SZ_GS  = (size_t)N_ * 32 * 4;
static constexpr size_t WS_SV  = WS_GS + SZ_GS;                      // f32 S_V [N*C][T]
static constexpr size_t SZ_SV  = (size_t)N_ * C_ * T_ * 4;
static constexpr size_t WS_GT  = WS_SV + SZ_SV;                      // f32 gt [N][T]
static constexpr size_t SZ_GT  = (size_t)N_ * T_ * 4;
static constexpr size_t WS_SEC = WS_GT + SZ_GT;                      // f32 se_c [N][C]
static constexpr size_t SZ_SEC = (size_t)N_ * C_ * 4;
static constexpr size_t WS_GC  = WS_SEC + SZ_SEC;                    // f32 gc [N][C]

// ---- WMMA fragment helpers (layouts per CDNA5 ISA 7.12.2) ----
// A 16x32 f16: row M = lane&15; K elements: e<8 -> g*8+e ; e>=8 -> 16+g*8+(e-8)
__device__ __forceinline__ v16h frag_a(const _Float16* row_k0, int g) {
    union { v16h v; v8h h[2]; } u;
    u.h[0] = *(const v8h*)(row_k0 + g * 8);
    u.h[1] = *(const v8h*)(row_k0 + 16 + g * 8);
    return u.v;
}
// B 32x16 f16: col N = lane&15; K elements: e<8 -> g*8+e ; e>=8 -> 16+g*8+(e-8) (storage: per-column contiguous in K)
__device__ __forceinline__ v16h frag_b(const _Float16* col_k0, int g) {
    union { v16h v; v8h h[2]; } u;
    u.h[0] = *(const v8h*)(col_k0 + g * 8);
    u.h[1] = *(const v8h*)(col_k0 + 16 + g * 8);
    return u.v;
}
__device__ __forceinline__ v8f wmma16(v16h a, v16h b, v8f c) {
    return __builtin_amdgcn_wmma_f32_16x16x32_f16(false, a, false, b, (short)0, c, false, false);
}
__device__ __forceinline__ float sigmoidf_(float x) { return 1.0f / (1.0f + expf(-x)); }

// ---------------------------------------------------------------------------
// K1: adaptive adjacency  AiT[s][n][w][v] = PA + tanh((a^T b)/4096)*alpha  (f16, zero-padded 32x32)
// one block per (s,n); 256 threads (8 waves)
// ---------------------------------------------------------------------------
__global__ __launch_bounds__(256) void adj_kernel(
    const float* __restrict__ x, const float* __restrict__ PA,
    const float* __restrict__ alpha,
    const float* __restrict__ Wa, const float* __restrict__ ba,
    const float* __restrict__ Wb, const float* __restrict__ bb,
    _Float16* __restrict__ AiT)
{
    __shared__ _Float16 xsT[512 * 64];   // [p = t*32+v][c]  (B for GEMM1)
    __shared__ _Float16 wcat[32 * 64];   // [r][c]  rows 0-15 = Wa, 16-31 = Wb
    __shared__ _Float16 a_arr[32 * 256]; // [v][k = e*16+t]  (A rows for GEMM2)
    __shared__ _Float16 b_arr[32 * 256]; // [w][k]           (B cols for GEMM2)
    __shared__ float    bias_ab[32];
    __shared__ float    red[8 * 1024];

    const int tid  = threadIdx.x;
    const int wid  = tid >> 5;
    const int lane = tid & 31;
    const int g    = lane >> 4;
    const int nn   = lane & 15;
    const int n    = blockIdx.x & 63;
    const int s    = blockIdx.x >> 6;

    for (int idx = tid; idx < 2048; idx += 256) {
        int r = idx >> 6, c = idx & 63;
        float w = (r < 16) ? Wa[(s * 16 + r) * 64 + c] : Wb[(s * 16 + (r - 16)) * 64 + c];
        wcat[idx] = (_Float16)w;
    }
    if (tid < 32) bias_ab[tid] = (tid < 16) ? ba[s * 16 + tid] : bb[s * 16 + (tid - 16)];

    v8f accA[4];
    for (int i = 0; i < 4; ++i) accA[i] = {};

    const float* xb = x + (size_t)n * CTV_;

    for (int tc = 0; tc < 16; ++tc) {
        __syncthreads();
        const int t0 = tc * 16;
        // stage x chunk (f32 -> f16), transposed [p][c], v>=25 zero-padded
        for (int p = tid; p < 512; p += 256) {
            int t = p >> 5, v = p & 31;
            if (v < V_) {
                const float* xr = xb + (size_t)(t0 + t) * V_ + v;
                for (int c = 0; c < 64; ++c) xsT[p * 64 + c] = (_Float16)xr[(size_t)c * TV_];
            } else {
                for (int c = 0; c < 64; ++c) xsT[p * 64 + c] = (_Float16)0.0f;
            }
        }
        __syncthreads();
        // GEMM1: ab[r][t,v] = Wcat @ x  (M=32 rows, K=64, N=512 positions)
        for (int idx = wid; idx < 64; idx += 8) {
            int mt = idx >> 5, pt = idx & 31;
            v8f d = {};
            const _Float16* arow = wcat + (mt * 16 + nn) * 64;
            const _Float16* bcol = xsT + (pt * 16 + nn) * 64;
            d = wmma16(frag_a(arow, g),      frag_b(bcol, g),      d);
            d = wmma16(frag_a(arow + 32, g), frag_b(bcol + 32, g), d);
            int p = pt * 16 + nn;
            int t = p >> 5, v = p & 31;
            for (int j = 0; j < 8; ++j) {
                int r = mt * 16 + j + 8 * g;
                float val = d[j] + bias_ab[r];
                if (r < 16) a_arr[v * 256 + r * 16 + t]        = (_Float16)val;
                else        b_arr[v * 256 + (r - 16) * 16 + t] = (_Float16)val;
            }
        }
        __syncthreads();
        // GEMM2: A1 += a^T b ; wave wid owns K window [wid*32, wid*32+32)
        {
            const int kb = wid * 32;
            for (int tt = 0; tt < 4; ++tt) {
                int vt = tt >> 1, wt = tt & 1;
                v16h af = frag_a(a_arr + (vt * 16 + nn) * 256 + kb, g);
                v16h bf = frag_b(b_arr + (wt * 16 + nn) * 256 + kb, g);
                accA[tt] = wmma16(af, bf, accA[tt]);
            }
        }
    }
    __syncthreads();
    // deterministic cross-wave reduce + epilogue
    for (int tt = 0; tt < 4; ++tt)
        for (int j = 0; j < 8; ++j)
            red[wid * 1024 + tt * 256 + (j + 8 * g) * 16 + nn] = accA[tt][j];
    __syncthreads();
    const float al = alpha[0];
    for (int idx = tid; idx < 1024; idx += 256) {
        float sum = 0.0f;
        for (int w = 0; w < 8; ++w) sum += red[w * 1024 + idx];
        int tt = idx >> 8, m = (idx >> 4) & 15, col = idx & 15;
        int v  = (tt >> 1) * 16 + m;
        int wc = (tt & 1) * 16 + col;
        float outv = 0.0f;
        if (v < V_ && wc < V_)
            outv = PA[s * (V_ * V_) + v * V_ + wc] + tanhf(sum * (1.0f / 4096.0f)) * al;
        AiT[((size_t)(s * 64 + n)) * 1024 + wc * 32 + v] = (_Float16)outv;
    }
}

// ---------------------------------------------------------------------------
// K2: fused y = BN(sum_s Wd[s] @ (x @ Ai[s]) + bd[s]) + x, ReLU
// block = (t-chunk of 8, n); 256 threads (8 waves)
// ---------------------------------------------------------------------------
__global__ __launch_bounds__(256) void gcn_kernel(
    const float* __restrict__ x, const _Float16* __restrict__ AiT,
    const float* __restrict__ Wd, const float* __restrict__ bd,
    const float* __restrict__ gamma, const float* __restrict__ beta,
    float* __restrict__ out)
{
    __shared__ _Float16 xs[512 * 32];    // [m = c*8+t][v]  (A for z-GEMM)
    __shared__ _Float16 zs[256 * 64];    // [p = t*32+w][c] (B for y-GEMM)
    __shared__ _Float16 wds[64 * 64];    // Wd[s] row-major [o][c]
    __shared__ _Float16 ais[32 * 32];    // AiT [w][v]      (B for z-GEMM)
    __shared__ float    ep_scale[64], ep_beta[64], ep_bsum[64];

    const int tid  = threadIdx.x;
    const int wid  = tid >> 5;
    const int lane = tid & 31;
    const int g    = lane >> 4;
    const int nn   = lane & 15;
    const int n    = blockIdx.y;
    const int t0   = blockIdx.x * 8;

    if (tid < 64) {
        ep_scale[tid] = gamma[tid] * rsqrtf(1.0f + 1e-5f);
        ep_beta[tid]  = beta[tid];
        ep_bsum[tid]  = bd[tid] + bd[64 + tid] + bd[128 + tid];
    }
    const float* xb = x + (size_t)n * CTV_;
    for (int idx = tid; idx < 512 * 32; idx += 256) {
        int m = idx >> 5, v = idx & 31;
        int c = m >> 3, t = m & 7;
        xs[idx] = (v < V_) ? (_Float16)xb[(size_t)c * TV_ + (size_t)(t0 + t) * V_ + v]
                           : (_Float16)0.0f;
    }

    v8f acc[8];
    for (int i = 0; i < 8; ++i) acc[i] = {};

    for (int s = 0; s < S_; ++s) {
        __syncthreads();  // protects zs/ais/wds reuse across subsets
        for (int idx = tid; idx < 1024; idx += 256)
            ais[idx] = AiT[((size_t)(s * 64 + n)) * 1024 + idx];
        for (int idx = tid; idx < 4096; idx += 256)
            wds[idx] = (_Float16)Wd[s * 4096 + idx];
        __syncthreads();
        // z-GEMM: z[(c,t)][w] = x[(c,t)][v] * Ai[v][w]   (K = 32, single WMMA)
        for (int idx = wid; idx < 64; idx += 8) {
            int mt = idx >> 1, wt = idx & 1;
            v8f d = {};
            d = wmma16(frag_a(xs + (mt * 16 + nn) * 32, g),
                       frag_b(ais + (wt * 16 + nn) * 32, g), d);
            for (int j = 0; j < 8; ++j) {
                int m = mt * 16 + j + 8 * g;
                int c = m >> 3, t = m & 7;
                zs[(t * 32 + wt * 16 + nn) * 64 + c] = (_Float16)d[j];
            }
        }
        __syncthreads();
        // y-GEMM: y[o][p] += Wd[o][c] * z[c][p]   (K = 64 -> 2 WMMAs)
        for (int a = 0; a < 8; ++a) {
            int id = wid * 8 + a;
            int ot = id & 3, pt = id >> 2;
            const _Float16* arow = wds + (ot * 16 + nn) * 64;
            const _Float16* bcol = zs + (pt * 16 + nn) * 64;
            acc[a] = wmma16(frag_a(arow, g),      frag_b(bcol, g),      acc[a]);
            acc[a] = wmma16(frag_a(arow + 32, g), frag_b(bcol + 32, g), acc[a]);
        }
    }
    __syncthreads();
    // epilogue: bias -> BN -> +residual -> ReLU -> store
    float* ob = out + (size_t)n * CTV_;
    for (int a = 0; a < 8; ++a) {
        int id = wid * 8 + a;
        int ot = id & 3, pt = id >> 2;
        int t = t0 + (pt >> 1);
        int v = (pt & 1) * 16 + nn;
        if (v < V_) {
            for (int j = 0; j < 8; ++j) {
                int o = ot * 16 + j + 8 * g;
                float val = acc[a][j] + ep_bsum[o];
                val = val * ep_scale[o] + ep_beta[o];
                val += xb[(size_t)o * TV_ + (size_t)t * V_ + v];
                ob[(size_t)o * TV_ + (size_t)t * V_ + v] = fmaxf(val, 0.0f);
            }
        }
    }
}

// ---------------------------------------------------------------------------
// K3: S_T[n,c,v] = mean_t y   (block per (n,c))
// ---------------------------------------------------------------------------
__global__ __launch_bounds__(256) void reduceT_kernel(const float* __restrict__ y,
                                                      float* __restrict__ S_T)
{
    __shared__ float red[256];
    const int nc  = blockIdx.x;
    const int tid = threadIdx.x;
    const int grp = tid >> 5, v = tid & 31;
    const float* yb = y + (size_t)nc * TV_;
    float p = 0.0f;
    if (v < V_)
        for (int t = grp; t < T_; t += 8) p += yb[t * V_ + v];
    red[tid] = p;
    __syncthreads();
    if (grp == 0 && v < V_) {
        float s = 0.0f;
        for (int gg = 0; gg < 8; ++gg) s += red[gg * 32 + v];
        S_T[(size_t)nc * 32 + v] = s * (1.0f / 256.0f);
    }
}

// K4: spatial gate  gs[n,v] = 1 + sigmoid(conv25_V(S_T))
__global__ __launch_bounds__(32) void gate_s_kernel(const float* __restrict__ S_T,
                                                    const float* __restrict__ Wsa,
                                                    const float* __restrict__ bsa,
                                                    float* __restrict__ gs)
{
    const int n = blockIdx.x, v = threadIdx.x;
    if (v >= 32) return;
    float sum = bsa[0];
    if (v < V_) {
        for (int c = 0; c < C_; ++c) {
            const float* st = S_T + ((size_t)n * C_ + c) * 32;
            const float* wr = Wsa + c * V_;
            for (int k = 0; k < V_; ++k) {
                int iv = v + k - 12;
                if (iv >= 0 && iv < V_) sum += wr[k] * st[iv];
            }
        }
        gs[n * 32 + v] = 1.0f + sigmoidf_(sum);
    } else {
        gs[n * 32 + v] = 0.0f;
    }
}

// K5: y *= gs[v]; S_V[n,c,t] = mean_v y'   (block per (n,c))
__global__ __launch_bounds__(256) void apply_s_kernel(float* __restrict__ y,
                                                      const float* __restrict__ gs,
                                                      float* __restrict__ S_V)
{
    const int nc  = blockIdx.x;
    const int tid = threadIdx.x;
    const int grp = tid >> 5, lane = tid & 31;
    const int n = nc >> 6;
    float* yb = y + (size_t)nc * TV_;
    const float gv = (lane < V_) ? gs[n * 32 + lane] : 0.0f;
    for (int t = grp; t < T_; t += 8) {
        float r = 0.0f;
        if (lane < V_) {
            float val = yb[t * V_ + lane] * gv;
            yb[t * V_ + lane] = val;
            r = val;
        }
        for (int off = 16; off; off >>= 1) r += __shfl_xor(r, off, 32);
        if (lane == 0) S_V[(size_t)nc * T_ + t] = r * (1.0f / 25.0f);
    }
}

// K6: temporal gate  gt[n,t] = 1 + sigmoid(conv9_T(S_V))
__global__ __launch_bounds__(256) void gate_t_kernel(const float* __restrict__ S_V,
                                                     const float* __restrict__ Wta,
                                                     const float* __restrict__ bta,
                                                     float* __restrict__ gt)
{
    const int n = blockIdx.x, t = threadIdx.x;
    float sum = bta[0];
    for (int c = 0; c < C_; ++c) {
        const float* sv = S_V + ((size_t)n * C_ + c) * T_;
        const float* wr = Wta + c * 9;
        for (int k = 0; k < 9; ++k) {
            int it = t + k - 4;
            if (it >= 0 && it < T_) sum += wr[k] * sv[it];
        }
    }
    gt[n * T_ + t] = 1.0f + sigmoidf_(sum);
}

// K7: y *= gt[t]; se_c[n,c] = mean_{t,v} y'   (block per (n,c))
__global__ __launch_bounds__(256) void apply_t_kernel(float* __restrict__ y,
                                                      const float* __restrict__ gt,
                                                      float* __restrict__ se_c)
{
    __shared__ float red8[8];
    const int nc  = blockIdx.x;
    const int tid = threadIdx.x;
    const int grp = tid >> 5, lane = tid & 31;
    const int n = nc >> 6;
    float* yb = y + (size_t)nc * TV_;
    float accum = 0.0f;
    for (int t = grp; t < T_; t += 8) {
        const float gv = gt[n * T_ + t];
        float r = 0.0f;
        if (lane < V_) {
            float val = yb[t * V_ + lane] * gv;
            yb[t * V_ + lane] = val;
            r = val;
        }
        accum += r;
    }
    for (int off = 16; off; off >>= 1) accum += __shfl_xor(accum, off, 32);
    if (lane == 0) red8[grp] = accum;
    __syncthreads();
    if (tid == 0) {
        float s = 0.0f;
        for (int gg = 0; gg < 8; ++gg) s += red8[gg];
        se_c[nc] = s * (1.0f / 6400.0f);
    }
}

// K8: channel gate  gc[n,c] = 1 + sigmoid(W2 @ relu(W1 @ se + b1) + b2)
__global__ __launch_bounds__(64) void gate_c_kernel(const float* __restrict__ se_c,
                                                    const float* __restrict__ W1,
                                                    const float* __restrict__ b1,
                                                    const float* __restrict__ W2,
                                                    const float* __restrict__ b2,
                                                    float* __restrict__ gc)
{
    __shared__ float seL[64];
    __shared__ float h[32];
    const int n = blockIdx.x, tid = threadIdx.x;
    seL[tid] = se_c[n * 64 + tid];
    __syncthreads();
    if (tid < 32) {
        float hh = b1[tid];
        for (int c = 0; c < 64; ++c) hh += W1[tid * 64 + c] * seL[c];
        h[tid] = fmaxf(hh, 0.0f);
    }
    __syncthreads();
    float s2 = b2[tid];
    for (int j = 0; j < 32; ++j) s2 += W2[tid * 32 + j] * h[j];
    gc[n * 64 + tid] = 1.0f + sigmoidf_(s2);
}

// K9: y *= gc[n,c]  (vectorized; 6400 % 4 == 0 so a float4 never crosses (n,c))
__global__ __launch_bounds__(256) void apply_c_kernel(float* __restrict__ y,
                                                      const float* __restrict__ gc)
{
    size_t e0 = ((size_t)blockIdx.x * 256 + threadIdx.x) * 4;
    const float gv = gc[e0 / TV_];
    float4 v = *(float4*)(y + e0);
    v.x *= gv; v.y *= gv; v.z *= gv; v.w *= gv;
    *(float4*)(y + e0) = v;
}

// ---------------------------------------------------------------------------
extern "C" void kernel_launch(void* const* d_in, const int* in_sizes, int n_in,
                              void* d_out, int out_size, void* d_ws, size_t ws_size,
                              hipStream_t stream) {
    (void)in_sizes; (void)n_in; (void)out_size; (void)ws_size;
    const float* x     = (const float*)d_in[0];
    const float* PA    = (const float*)d_in[1];
    const float* alpha = (const float*)d_in[2];
    const float* Wa    = (const float*)d_in[3];
    const float* ba    = (const float*)d_in[4];
    const float* Wb    = (const float*)d_in[5];
    const float* bb    = (const float*)d_in[6];
    const float* Wd    = (const float*)d_in[7];
    const float* bd    = (const float*)d_in[8];
    const float* gamma = (const float*)d_in[9];
    const float* beta  = (const float*)d_in[10];
    const float* Wsa   = (const float*)d_in[11];
    const float* bsa   = (const float*)d_in[12];
    const float* Wta   = (const float*)d_in[13];
    const float* bta   = (const float*)d_in[14];
    const float* W1    = (const float*)d_in[15];
    const float* b1    = (const float*)d_in[16];
    const float* W2    = (const float*)d_in[17];
    const float* b2    = (const float*)d_in[18];
    float* out = (float*)d_out;
    char*  ws  = (char*)d_ws;

    _Float16* AiT = (_Float16*)(ws + WS_AIT);
    float* S_T  = (float*)(ws + WS_ST);
    float* GS   = (float*)(ws + WS_GS);
    float* S_V  = (float*)(ws + WS_SV);
    float* GT   = (float*)(ws + WS_GT);
    float* SEC  = (float*)(ws + WS_SEC);
    float* GC   = (float*)(ws + WS_GC);

    adj_kernel<<<S_ * N_, 256, 0, stream>>>(x, PA, alpha, Wa, ba, Wb, bb, AiT);
    gcn_kernel<<<dim3(T_ / 8, N_), 256, 0, stream>>>(x, AiT, Wd, bd, gamma, beta, out);
    reduceT_kernel<<<N_ * C_, 256, 0, stream>>>(out, S_T);
    gate_s_kernel<<<N_, 32, 0, stream>>>(S_T, Wsa, bsa, GS);
    apply_s_kernel<<<N_ * C_, 256, 0, stream>>>(out, GS, S_V);
    gate_t_kernel<<<N_, 256, 0, stream>>>(S_V, Wta, bta, GT);
    apply_t_kernel<<<N_ * C_, 256, 0, stream>>>(out, GT, SEC);
    gate_c_kernel<<<N_, 64, 0, stream>>>(SEC, W1, b1, W2, b2, GC);
    apply_c_kernel<<<(N_ * C_ * T_ * V_) / 1024, 256, 0, stream>>>(out, GC);
}
